// LinearAttention_1606317768998
// MI455X (gfx1250) — matrix operand order repaired and checked by reading, not compiled, not run on hardware
//
#include <hip/hip_runtime.h>
#include <hip/hip_bf16.h>
#include <stdint.h>

// ---------------------------------------------------------------------------
// CDNA5 (gfx1250) linear-attention pipeline, bf16 WMMA (16x16x32, f32 acc).
// ---------------------------------------------------------------------------

typedef __attribute__((ext_vector_type(16))) __bf16 bf16x16;
typedef __attribute__((ext_vector_type(8)))  float  f32x8;

union FragAB { bf16x16 v; uint32_t u[8]; };

#define NORM_EPS 1e-5f

// ---------------------------------------------------------------------------
// Generic TN GEMM:  C[M,N] = A[M,K] * Bt[N,K]^T   (all row-major, bf16 in,
// f32 accumulate, f32 or bf16 out).  Each wave computes a 32x32 output tile:
// 2 A frags x 2 B frags -> 4 v_wmma_f32_16x16x32_bf16 per K-step, every
// fragment reused twice.  8 waves (4 along M, 2 along N) tile a 128x64 block.
// Requires M % 128 == 0, N % 64 == 0, K % 32 == 0.
// ---------------------------------------------------------------------------
template <bool OUT_BF16>
__global__ __launch_bounds__(256) void gemm_bf16_tn(
    const __bf16* __restrict__ A,  long long sA,   // batch stride (elements)
    const __bf16* __restrict__ Bt, long long sB,
    void* __restrict__ Cv,         long long sC,
    int M, int N, int K)
{
  const int bz = blockIdx.z;
  A  += (long long)bz * sA;
  Bt += (long long)bz * sB;

  const int lane = threadIdx.x & 31;
  const int wave = threadIdx.x >> 5;
  const int half = lane >> 4;     // K-half selector (ISA 16-bit layouts)
  const int ml   = lane & 15;     // row (A) / col (B,C) within tile

  const int wm = wave & 3;        // 4 waves along M
  const int wn = wave >> 2;       // 2 waves along N
  const int m0 = blockIdx.y * 128 + wm * 32;
  const int n0 = blockIdx.x * 64 + wn * 32;

  const __bf16* Arow0 = A  + (long long)(m0 + ml) * K;
  const __bf16* Arow1 = Arow0 + (long long)16 * K;
  const __bf16* Brow0 = Bt + (long long)(n0 + ml) * K;
  const __bf16* Brow1 = Brow0 + (long long)16 * K;

  f32x8 acc00 = {0.f, 0.f, 0.f, 0.f, 0.f, 0.f, 0.f, 0.f};
  f32x8 acc01 = acc00, acc10 = acc00, acc11 = acc00;

  for (int k0 = 0; k0 < K; k0 += 32) {
    FragAB a0, a1, b0, b1;
#pragma unroll
    for (int r = 0; r < 8; ++r) {
      // A 16x32 bf16 layout: pair e=2r holds K = 2r + 8*((r>=4) + half)
      const int ka = k0 + 2 * r + 8 * (((r >> 2) & 1) + half);
      a0.u[r] = *(const uint32_t*)(Arow0 + ka);
      a1.u[r] = *(const uint32_t*)(Arow1 + ka);
      // B 32x16 bf16 layout: pair e=2r holds K = 2r + 8*((r>=4) + half)
      const int kb = k0 + 2 * r + 8 * (((r >> 2) & 1) + half);
      b0.u[r] = *(const uint32_t*)(Brow0 + kb);
      b1.u[r] = *(const uint32_t*)(Brow1 + kb);
    }
    acc00 = __builtin_amdgcn_wmma_f32_16x16x32_bf16(false, a0.v, false, b0.v,
                                                    (short)0, acc00, false, false);
    acc01 = __builtin_amdgcn_wmma_f32_16x16x32_bf16(false, a0.v, false, b1.v,
                                                    (short)0, acc01, false, false);
    acc10 = __builtin_amdgcn_wmma_f32_16x16x32_bf16(false, a1.v, false, b0.v,
                                                    (short)0, acc10, false, false);
    acc11 = __builtin_amdgcn_wmma_f32_16x16x32_bf16(false, a1.v, false, b1.v,
                                                    (short)0, acc11, false, false);
  }

  // C/D 16x16 f32 layout: VGPR r -> row (r + 8*half), col = ml
  if (OUT_BF16) {
    __bf16* C = (__bf16*)Cv + (long long)bz * sC;
#pragma unroll
    for (int r = 0; r < 8; ++r) {
      const long long row0 = (long long)(m0 + r + 8 * half) * N;
      const long long row1 = row0 + (long long)16 * N;
      C[row0 + n0 + ml]      = (__bf16)acc00[r];
      C[row0 + n0 + 16 + ml] = (__bf16)acc01[r];
      C[row1 + n0 + ml]      = (__bf16)acc10[r];
      C[row1 + n0 + 16 + ml] = (__bf16)acc11[r];
    }
  } else {
    float* C = (float*)Cv + (long long)bz * sC;
#pragma unroll
    for (int r = 0; r < 8; ++r) {
      const long long row0 = (long long)(m0 + r + 8 * half) * N;
      const long long row1 = row0 + (long long)16 * N;
      C[row0 + n0 + ml]      = acc00[r];
      C[row0 + n0 + 16 + ml] = acc01[r];
      C[row1 + n0 + ml]      = acc10[r];
      C[row1 + n0 + 16 + ml] = acc11[r];
    }
  }
}

// ---------------------------------------------------------------------------
// f32 -> bf16 elementwise convert
// ---------------------------------------------------------------------------
__global__ void cvt_f32_bf16(const float* __restrict__ in,
                             __bf16* __restrict__ out, long long n)
{
  long long i = (long long)blockIdx.x * blockDim.x + threadIdx.x;
  if (i < n) out[i] = (__bf16)in[i];
}

// ---------------------------------------------------------------------------
// Transpose + convert weight matrix: in f32 [R,C] -> out bf16 [C,R]
// ---------------------------------------------------------------------------
__global__ void transpose_cvt_w(const float* __restrict__ in,
                                __bf16* __restrict__ out, int R, int C)
{
  int i = blockIdx.x * blockDim.x + threadIdx.x;
  if (i < R * C) {
    int r = i / C, c = i % C;
    out[(long long)c * R + r] = (__bf16)in[i];
  }
}

// ---------------------------------------------------------------------------
// Weighted per-channel stats over tokens: mean = sum_n w*x,
// invstd = rsqrt(sum_n w*x^2 - mean^2 + eps)  (sum w == 1).
// X: bf16 [B,N,D].  block (64,4), grid (D/64, B).
// ---------------------------------------------------------------------------
__global__ __launch_bounds__(256) void weighted_stats(
    const __bf16* __restrict__ X, const float* __restrict__ W,
    float* __restrict__ mean, float* __restrict__ invstd, int N, int D)
{
  const int b = blockIdx.y;
  const int d = blockIdx.x * 64 + threadIdx.x;
  const __bf16* Xb = X + (long long)b * N * D;
  const float*  Wb = W + (long long)b * N;

  float s1 = 0.f, s2 = 0.f;
  for (int n = threadIdx.y; n < N; n += 4) {
    float w = Wb[n];
    float x = (float)Xb[(long long)n * D + d];
    s1 += w * x;
    s2 += w * x * x;
  }
  __shared__ float sh1[4][64], sh2[4][64];
  sh1[threadIdx.y][threadIdx.x] = s1;
  sh2[threadIdx.y][threadIdx.x] = s2;
  __syncthreads();
  if (threadIdx.y == 0) {
    s1 = sh1[0][threadIdx.x] + sh1[1][threadIdx.x] +
         sh1[2][threadIdx.x] + sh1[3][threadIdx.x];
    s2 = sh2[0][threadIdx.x] + sh2[1][threadIdx.x] +
         sh2[2][threadIdx.x] + sh2[3][threadIdx.x];
    float m   = s1;
    float var = s2 - m * m;
    mean[(long long)b * D + d]   = m;
    invstd[(long long)b * D + d] = rsqrtf(var + NORM_EPS);
  }
}

// ---------------------------------------------------------------------------
// Normalize (+ scale/bias, optional *weights) and transpose [B,N,D]->[B,D,N]
// via LDS 32x33 tile.  block (32,8), grid (D/32, N/32, B).
// ---------------------------------------------------------------------------
template <bool MUL_W>
__global__ __launch_bounds__(256) void norm_transpose(
    const __bf16* __restrict__ X, const float* __restrict__ W,
    const float* __restrict__ mean, const float* __restrict__ invstd,
    const float* __restrict__ scale, const float* __restrict__ bias,
    __bf16* __restrict__ Y, int N, int D)
{
  __shared__ float tile[32][33];
  const int b  = blockIdx.z;
  const int d0 = blockIdx.x * 32;
  const int n0 = blockIdx.y * 32;
  const int tx = threadIdx.x, ty = threadIdx.y;

  const int   d  = d0 + tx;
  const float is = invstd[(long long)b * D + d];
  const float sc = scale[d] * is;
  const float bi = bias[d] - mean[(long long)b * D + d] * sc;

#pragma unroll
  for (int j = 0; j < 4; ++j) {
    const int n = n0 + ty + j * 8;
    float v = (float)X[((long long)b * N + n) * D + d] * sc + bi;
    if (MUL_W) v *= W[(long long)b * N + n];
    tile[ty + j * 8][tx] = v;           // tile[n_local][d_local]
  }
  __syncthreads();
#pragma unroll
  for (int j = 0; j < 4; ++j) {
    const int d2 = d0 + ty + j * 8;
    const int n2 = n0 + tx;
    Y[((long long)b * D + d2) * N + n2] = (__bf16)tile[tx][ty + j * 8];
  }
}

// ---------------------------------------------------------------------------
// Host orchestration
// ---------------------------------------------------------------------------
extern "C" void kernel_launch(void* const* d_in, const int* in_sizes, int n_in,
                              void* d_out, int out_size, void* d_ws, size_t ws_size,
                              hipStream_t stream)
{
  (void)n_in; (void)ws_size; (void)out_size;
  const int B = 8, Ntok = 4096;
  const int D = in_sizes[8];                 // key_scale: [D] -> 512
  const int E = in_sizes[4] / D;             // Wq: [E,D]  -> 512

  const float* query  = (const float*)d_in[0];
  const float* key    = (const float*)d_in[1];
  const float* value  = (const float*)d_in[2];
  const float* wts    = (const float*)d_in[3];
  const float* Wq     = (const float*)d_in[4];
  const float* Wk     = (const float*)d_in[5];
  const float* Wv     = (const float*)d_in[6];
  const float* Wo     = (const float*)d_in[7];
  const float* kscale = (const float*)d_in[8];
  const float* kbias  = (const float*)d_in[9];
  const float* vscale = (const float*)d_in[10];
  const float* vbias  = (const float*)d_in[11];

  // ---- workspace layout (bytes) ----
  char* ws = (char*)d_ws;
  const size_t szND2 = (size_t)B * Ntok * D * 2;       // 32 MB per [B,N,D] bf16
  __bf16* inbuf = (__bf16*)(ws + 0 * szND2);           // reused q/k/v bf16 input
  __bf16* k_nd  = (__bf16*)(ws + 1 * szND2);           // k proj [B,N,D]
  __bf16* v_nd  = (__bf16*)(ws + 2 * szND2);           // v proj [B,N,D]
  __bf16* q_bf  = (__bf16*)(ws + 3 * szND2);           // q proj [B,N,D]
  __bf16* k_dn  = (__bf16*)(ws + 4 * szND2);           // normed*w k [B,D,N]
  __bf16* v_dn  = (__bf16*)(ws + 5 * szND2);           // normed   v [B,D,N]
  char*   p     = ws + 6 * szND2;
  __bf16* kvT   = (__bf16*)p;  p += (size_t)B * D * D * 2;   // kv^T [B,D,D]
  __bf16* Wq_t  = (__bf16*)p;  p += (size_t)D * E * 2;
  __bf16* Wk_t  = (__bf16*)p;  p += (size_t)D * E * 2;
  __bf16* Wv_t  = (__bf16*)p;  p += (size_t)D * E * 2;
  __bf16* Wo_t  = (__bf16*)p;  p += (size_t)E * D * 2;
  float*  kmean = (float*)p;   p += (size_t)B * D * 4;
  float*  kinv  = (float*)p;   p += (size_t)B * D * 4;
  float*  vmean = (float*)p;   p += (size_t)B * D * 4;
  float*  vinv  = (float*)p;
  __bf16* attn  = k_nd;        // reuse: k_nd dead after norm_transpose

  const long long nIn = (long long)B * Ntok * E;
  const long long sIn = (long long)Ntok * E;   // batch strides (elements)
  const long long sND = (long long)Ntok * D;
  const long long sDN = (long long)D * Ntok;
  const long long sDD = (long long)D * D;

  const dim3 blk256(256);
  const dim3 gW((E * D + 255) / 256);
  const dim3 gCvt((unsigned)((nIn + 255) / 256));
  const dim3 gProj(D / 64, Ntok / 128, B);     // M=Ntok, N=D, K=E
  const dim3 gKV(D / 64, D / 128, B);          // M=D,    N=D, K=Ntok
  const dim3 gStat(D / 64, B);
  const dim3 bStat(64, 4);
  const dim3 gNT(D / 32, Ntok / 32, B);
  const dim3 bNT(32, 8);

  // 0) transpose-convert weights to [cols,K] bf16
  transpose_cvt_w<<<gW, blk256, 0, stream>>>(Wq, Wq_t, E, D);
  transpose_cvt_w<<<gW, blk256, 0, stream>>>(Wk, Wk_t, E, D);
  transpose_cvt_w<<<gW, blk256, 0, stream>>>(Wv, Wv_t, E, D);
  transpose_cvt_w<<<gW, blk256, 0, stream>>>(Wo, Wo_t, D, E);

  // 1) k = key @ Wk   (bf16 out)
  cvt_f32_bf16<<<gCvt, blk256, 0, stream>>>(key, inbuf, nIn);
  gemm_bf16_tn<true><<<gProj, blk256, 0, stream>>>(inbuf, sIn, Wk_t, 0,
                                                   k_nd, sND, Ntok, D, E);
  // 2) v = value @ Wv
  cvt_f32_bf16<<<gCvt, blk256, 0, stream>>>(value, inbuf, nIn);
  gemm_bf16_tn<true><<<gProj, blk256, 0, stream>>>(inbuf, sIn, Wv_t, 0,
                                                   v_nd, sND, Ntok, D, E);
  // 3) q = query @ Wq
  cvt_f32_bf16<<<gCvt, blk256, 0, stream>>>(query, inbuf, nIn);
  gemm_bf16_tn<true><<<gProj, blk256, 0, stream>>>(inbuf, sIn, Wq_t, 0,
                                                   q_bf, sND, Ntok, D, E);

  // 4) weighted instance-norm stats for k and v
  weighted_stats<<<gStat, bStat, 0, stream>>>(k_nd, wts, kmean, kinv, Ntok, D);
  weighted_stats<<<gStat, bStat, 0, stream>>>(v_nd, wts, vmean, vinv, Ntok, D);

  // 5) normalize (+ k *= weights) and transpose to [B,D,N]
  norm_transpose<true><<<gNT, bNT, 0, stream>>>(k_nd, wts, kmean, kinv,
                                                kscale, kbias, k_dn, Ntok, D);
  norm_transpose<false><<<gNT, bNT, 0, stream>>>(v_nd, wts, vmean, vinv,
                                                 vscale, vbias, v_dn, Ntok, D);

  // 6) kvT[e,d] = sum_n v[n,e]*k[n,d]   (M=D, N=D, K=Ntok; L2-resident)
  gemm_bf16_tn<true><<<gKV, blk256, 0, stream>>>(v_dn, sDN, k_dn, sDN,
                                                 kvT, sDD, D, D, Ntok);

  // 7) attn[n,e] = sum_d q[n,d]*kv[d,e]  (Bt = kvT)
  gemm_bf16_tn<true><<<gProj, blk256, 0, stream>>>(q_bf, sND, kvT, sDD,
                                                   attn, sND, Ntok, D, D);

  // 8) out = attn @ Wo  (f32 out straight to d_out)
  gemm_bf16_tn<false><<<gProj, blk256, 0, stream>>>(attn, sND, Wo_t, 0,
                                                    (float*)d_out, sIn,
                                                    Ntok, E, D);
}
